// MultiHeadAttn_16681652978122
// MI455X (gfx1250) — matrix operand order repaired and checked by reading, not compiled, not run on hardware
//
#include <hip/hip_runtime.h>
#include <math.h>

typedef __bf16 bf16;
typedef __attribute__((ext_vector_type(16))) __bf16 v16bf;
typedef __attribute__((ext_vector_type(8)))  __bf16 v8bf;
typedef __attribute__((ext_vector_type(8)))  float  v8f;

#define SEQ    1024
#define BATCH  2
#define DMODEL 1024
#define NHEAD  16
#define DHEAD  64
#define MROWS  (SEQ*BATCH)   // 2048

// -------- fragment helpers (this code's contiguous k order k = 16*kg + s, same for A and B) --------
// A (16x32 bf16): lane(0..15)=row M=lane, kg=lane>>4 selects K halves:
//   a[0..15] = A[row][kg*16 .. kg*16+15]
static __device__ __forceinline__ v16bf load_a_frag(const bf16* rowp, int kg) {
    return *((const v16bf*)(rowp + (kg << 4)));
}

static __device__ __forceinline__ v8f wmma_bf16(v16bf a, v16bf b, v8f c) {
    return __builtin_amdgcn_wmma_f32_16x16x32_bf16(false, a, false, b, (short)0, c, false, false);
}

// CDNA5 async DMA: 16B per lane, global -> LDS, tracked by ASYNCcnt.
// LDS aperture: low 32 bits of generic address == LDS byte offset (ISA 10.2).
static __device__ __forceinline__ void async_copy16(bf16* lds_dst, const bf16* gsrc) {
    unsigned lds = (unsigned)(unsigned long long)lds_dst;
    asm volatile("global_load_async_to_lds_b128 %0, %1, off"
                 :: "v"(lds), "v"(gsrc) : "memory");
}
static __device__ __forceinline__ void wait_async0() {
    asm volatile("s_wait_asynccnt 0x0" ::: "memory");
}

// ============ 128x128x32 double-buffered GEMM:  C = A * Bt^T ============
// A[M x K] bf16 row-major, Bt[N x K] bf16 row-major, 256 threads (8 waves).
// Wave grid 4(M:32 rows) x 2(N:64 cols); 8 WMMA tiles/wave/K-step.
// Async-DMA stages slab k+1 into the other LDS buffer while WMMAs run on k.
// All 6 fragments are loaded before the 8 back-to-back WMMAs (single dscnt wait).
template <typename Epi>
static __device__ void gemm128(const bf16* __restrict__ A, int lda,
                               const bf16* __restrict__ Bt, int ldb,
                               int K, int bm, int bn, Epi epi)
{
    __shared__ __attribute__((aligned(64))) bf16 As[2][128][32];
    __shared__ __attribute__((aligned(64))) bf16 Bs[2][128][32];
    const int t    = threadIdx.x;
    const int lane = t & 31;
    const int wave = t >> 5;
    const int wm   = wave & 3;        // 4 M-groups of 32 rows
    const int wn   = wave >> 2;       // 2 N-groups of 64 cols
    const int lr   = t >> 2;          // 0..63 (each thread stages 2 rows per tile)
    const int lc   = (t & 3) << 3;    // 0,8,16,24 (8 bf16 = 16B)
    const int kg   = lane >> 4;
    const int l15  = lane & 15;

    auto stage = [&](int k0, int buf) {
        async_copy16(&As[buf][lr][lc],      &A [(size_t)(bm + lr)      * lda + k0 + lc]);
        async_copy16(&As[buf][64 + lr][lc], &A [(size_t)(bm + 64 + lr) * lda + k0 + lc]);
        async_copy16(&Bs[buf][lr][lc],      &Bt[(size_t)(bn + lr)      * ldb + k0 + lc]);
        async_copy16(&Bs[buf][64 + lr][lc], &Bt[(size_t)(bn + 64 + lr) * ldb + k0 + lc]);
    };

    v8f acc[2][4] = {};
    stage(0, 0);
    wait_async0();
    __syncthreads();

    int buf = 0;
    for (int k0 = 0; k0 < K; k0 += 32) {
        if (k0 + 32 < K) stage(k0 + 32, buf ^ 1);   // async DMA overlaps WMMAs below
        // ---- load ALL fragments first (one LDS wait), then 8 WMMAs back-to-back
        v16bf a0 = load_a_frag(&As[buf][(wm << 5) + l15][0],      kg);
        v16bf a1 = load_a_frag(&As[buf][(wm << 5) + 16 + l15][0], kg);
        v16bf b[4];
#pragma unroll
        for (int nt = 0; nt < 4; ++nt)
            b[nt] = *(const v16bf*)&Bs[buf][(wn << 6) + (nt << 4) + l15][kg << 4];
#pragma unroll
        for (int nt = 0; nt < 4; ++nt) {
            acc[0][nt] = wmma_bf16(a0, b[nt], acc[0][nt]);
            acc[1][nt] = wmma_bf16(a1, b[nt], acc[1][nt]);
        }
        wait_async0();       // next buffer fully resident
        __syncthreads();     // everyone done reading current buffer
        buf ^= 1;
    }
    const int rb = bm + (wm << 5) + (kg << 3);
    const int cb = bn + (wn << 6) + l15;
#pragma unroll
    for (int mt = 0; mt < 2; ++mt)
#pragma unroll
        for (int nt = 0; nt < 4; ++nt)
#pragma unroll
            for (int i = 0; i < 8; ++i)
                epi(rb + (mt << 4) + i, cb + (nt << 4), acc[mt][nt][i]);
}

// ============ 64x64x32 GEMM (kept for N=64 case) ============
template <typename Epi>
static __device__ void gemm64x64(const bf16* __restrict__ A, int lda,
                                 const bf16* __restrict__ Bt, int ldb,
                                 int K, int bm, int bn, Epi epi)
{
    __shared__ __attribute__((aligned(64))) bf16 As[64][32];
    __shared__ __attribute__((aligned(64))) bf16 Bs[64][32];
    const int t    = threadIdx.x;
    const int lane = t & 31;
    const int wave = t >> 5;
    const int wm   = wave & 3;
    const int wn   = wave >> 2;
    const int lr   = t >> 2;
    const int lc   = (t & 3) << 3;
    const int kg   = lane >> 4;
    const int l15  = lane & 15;

    v8f acc0 = {}, acc1 = {};
    for (int k0 = 0; k0 < K; k0 += 32) {
        async_copy16(&As[lr][lc], &A [(size_t)(bm + lr) * lda + k0 + lc]);
        async_copy16(&Bs[lr][lc], &Bt[(size_t)(bn + lr) * ldb + k0 + lc]);
        if (k0 + 32 < K) {
            __builtin_prefetch(&A [(size_t)(bm + lr) * lda + k0 + 32 + lc], 0, 1);
            __builtin_prefetch(&Bt[(size_t)(bn + lr) * ldb + k0 + 32 + lc], 0, 1);
        }
        wait_async0();
        __syncthreads();
        v16bf a  = load_a_frag(&As[(wm << 4) + l15][0], kg);
        v16bf b0 = *(const v16bf*)&Bs[(wn << 5) + l15][kg << 4];
        v16bf b1 = *(const v16bf*)&Bs[(wn << 5) + 16 + l15][kg << 4];
        acc0 = wmma_bf16(a, b0, acc0);
        acc1 = wmma_bf16(a, b1, acc1);
        __syncthreads();
    }
    const int rbase = bm + (wm << 4) + (kg << 3);
    const int c0    = bn + (wn << 5) + l15;
#pragma unroll
    for (int i = 0; i < 8; ++i) epi(rbase + i, c0,      acc0[i]);
#pragma unroll
    for (int i = 0; i < 8; ++i) epi(rbase + i, c0 + 16, acc1[i]);
}

// ---------------- kernel 1: f32 -> bf16 packing ----------------
__global__ void __launch_bounds__(256) k_pack(const float* __restrict__ h,
        const float* __restrict__ Wq, const float* __restrict__ Wkv,
        const float* __restrict__ Wo,
        bf16* __restrict__ hb, bf16* __restrict__ Wqkv, bf16* __restrict__ Wob)
{
    size_t i = (size_t)blockIdx.x * 256 + threadIdx.x;     // 0 .. 2M-1
    const size_t M1 = 1024u * 1024u;
    if (i < 2 * M1) { hb[i] = (bf16)h[i]; Wqkv[M1 + i] = (bf16)Wkv[i]; }
    if (i < M1)     { Wqkv[i] = (bf16)Wq[i]; Wob[i] = (bf16)Wo[i]; }
}

// ---------------- kernel 2: QKV projection, scatter to per-head layouts ----
__global__ void __launch_bounds__(256) k_gemm_qkv(const bf16* __restrict__ hb,
        const bf16* __restrict__ Wqkv,
        bf16* __restrict__ Q, bf16* __restrict__ Kc, bf16* __restrict__ Vt)
{
    gemm128(hb, DMODEL, Wqkv, DMODEL, DMODEL, blockIdx.y << 7, blockIdx.x << 7,
        [=](int m, int n, float v) {
            int s = m >> 1, b = m & 1;
            bf16 x = (bf16)v;
            if (n < 1024) {                     // Q[bh][s][d]
                int hh = n >> 6, d = n & 63;
                Q[(((size_t)(b * NHEAD + hh)) * SEQ + s) * DHEAD + d] = x;
            } else if (n < 2048) {              // K[bh][s][d] (natural = Bt for scores)
                int nn = n - 1024, hh = nn >> 6, d = nn & 63;
                Kc[(((size_t)(b * NHEAD + hh)) * SEQ + s) * DHEAD + d] = x;
            } else {                            // V^T[bh][d][s] (= Bt for Pnew*V)
                int nn = n - 2048, hh = nn >> 6, d = nn & 63;
                Vt[(((size_t)(b * NHEAD + hh)) * DHEAD + d) * SEQ + s] = x;
            }
        });
}

// ---------------- kernel 3: scores + softmax, writes P and P^T -------------
__global__ void __launch_bounds__(256) k_attn(const bf16* __restrict__ Q,
        const bf16* __restrict__ Kc, bf16* __restrict__ P, bf16* __restrict__ Pt)
{
    extern __shared__ char smem[];
    float* sc    = (float*)smem;                     // [16][1024] f32 scores (64KB)
    bf16*  Qs    = (bf16*)(smem + 16 * SEQ * 4);     // [16][64] bf16 (2KB)
    float* red   = (float*)(smem + 16 * SEQ * 4);    // reuse after Qs dead: [16][16]
    float* rstat = red + 256;                        // [32]: rowmax, 1/rowsum

    const int t = threadIdx.x, lane = t & 31, wave = t >> 5;
    const int bh = blockIdx.y;
    const int i0 = blockIdx.x << 4;
    const bf16* Qg = Q  + ((size_t)bh * SEQ + i0) * DHEAD;
    const bf16* Kg = Kc + (size_t)bh * SEQ * DHEAD;

    if (t < 128) async_copy16(&Qs[t * 8], &Qg[t * 8]);
    wait_async0();
    __syncthreads();

    const int l15 = lane & 15, kg = lane >> 4;
    const int jw = wave << 7;                        // 128 cols per wave
    v8f acc[8] = {};
#pragma unroll
    for (int kk = 0; kk < DHEAD; kk += 32) {
        v16bf a = load_a_frag(&Qs[l15 * DHEAD + kk], kg);
#pragma unroll
        for (int t8 = 0; t8 < 8; ++t8) {
            int j = jw + (t8 << 4) + l15;
            v16bf b = *(const v16bf*)&Kg[(size_t)j * DHEAD + kk + (kg << 4)];
            acc[t8] = wmma_bf16(a, b, acc[t8]);
        }
    }
#pragma unroll
    for (int t8 = 0; t8 < 8; ++t8) {
        int c = jw + (t8 << 4) + l15;
#pragma unroll
        for (int i = 0; i < 8; ++i)
            sc[((kg << 3) + i) * SEQ + c] = acc[t8][i] * 0.125f;   // 1/sqrt(64)
    }
    __syncthreads();

    // softmax over j: 16 threads per row, 64 cols each
    const int r = t >> 4, seg = t & 15;
    float* rowp = sc + r * SEQ + seg * 64;
    float mx = -3.0e38f;
    for (int j = 0; j < 64; ++j) mx = fmaxf(mx, rowp[j]);
    red[r * 16 + seg] = mx;
    __syncthreads();
    if (t < 16) {
        float m2 = red[t * 16];
        for (int j2 = 1; j2 < 16; ++j2) m2 = fmaxf(m2, red[t * 16 + j2]);
        rstat[t] = m2;
    }
    __syncthreads();
    float rm = rstat[r], s = 0.f;
    for (int j = 0; j < 64; ++j) { float e = __expf(rowp[j] - rm); rowp[j] = e; s += e; }
    red[r * 16 + seg] = s;
    __syncthreads();
    if (t < 16) {
        float sum = 0.f;
        for (int j2 = 0; j2 < 16; ++j2) sum += red[t * 16 + j2];
        rstat[16 + t] = 1.0f / sum;
    }
    __syncthreads();
    float inv = rstat[16 + r];
    bf16* Pg  = P  + (size_t)bh * SEQ * SEQ;
    bf16* Ptg = Pt + (size_t)bh * SEQ * SEQ;
    for (int j = 0; j < 64; ++j) {
        int col = seg * 64 + j;
        bf16 pv = (bf16)(rowp[j] * inv);
        Pg [(size_t)(i0 + r) * SEQ + col] = pv;
        Ptg[(size_t)col * SEQ + (i0 + r)] = pv;
    }
}

// ---------------- kernel 4: twicing  Pnew = 2P - P@P  (batched 32x) --------
__global__ void __launch_bounds__(256) k_twice(const bf16* __restrict__ P,
        const bf16* __restrict__ Pt, bf16* __restrict__ Pn)
{
    size_t zo = (size_t)blockIdx.z * SEQ * SEQ;
    const bf16* Pb = P + zo; const bf16* Ptb = Pt + zo; bf16* Pnb = Pn + zo;
    gemm128(Pb, SEQ, Ptb, SEQ, SEQ, blockIdx.y << 7, blockIdx.x << 7,
        [=](int m, int n, float v) {
            float p0 = (float)Pb[(size_t)m * SEQ + n];
            Pnb[(size_t)m * SEQ + n] = (bf16)(2.0f * p0 - v);
        });
}

// ---------------- kernel 5: vec = Pnew @ V  (batched 32x, N=64) ------------
__global__ void __launch_bounds__(256) k_av(const bf16* __restrict__ Pn,
        const bf16* __restrict__ Vt, bf16* __restrict__ vec)
{
    int z = blockIdx.z, b = z >> 4, hh = z & 15;
    const bf16* A  = Pn + (size_t)z * SEQ * SEQ;
    const bf16* Bt = Vt + (size_t)z * DHEAD * SEQ;   // [64][1024]
    gemm64x64(A, SEQ, Bt, SEQ, SEQ, blockIdx.y << 6, 0,
        [=](int m, int n, float v) {
            vec[((size_t)m * BATCH + b) * (NHEAD * DHEAD) + hh * DHEAD + n] = (bf16)v;
        });
}

// ---------------- kernel 6: attn_out = vec @ Wo^T ; x = h + attn_out -------
__global__ void __launch_bounds__(256) k_out(const bf16* __restrict__ vec,
        const bf16* __restrict__ Wob, const float* __restrict__ h,
        float* __restrict__ xb)
{
    gemm128(vec, DMODEL, Wob, DMODEL, DMODEL, blockIdx.y << 7, blockIdx.x << 7,
        [=](int m, int n, float v) {
            size_t idx = (size_t)m * DMODEL + n;
            xb[idx] = h[idx] + v;
        });
}

// ---------------- kernel 7: LayerNorm --------------------------------------
__global__ void __launch_bounds__(256) k_ln(const float* __restrict__ xb,
        const float* __restrict__ gamma, const float* __restrict__ beta,
        float* __restrict__ out)
{
    __shared__ float red[256];
    const int m = blockIdx.x, t = threadIdx.x;
    const float* x = xb + (size_t)m * DMODEL;
    float s = 0.f;
    for (int j = t; j < DMODEL; j += 256) s += x[j];
    red[t] = s; __syncthreads();
    for (int o = 128; o > 0; o >>= 1) { if (t < o) red[t] += red[t + o]; __syncthreads(); }
    float mu = red[0] * (1.0f / DMODEL);
    __syncthreads();
    float v = 0.f;
    for (int j = t; j < DMODEL; j += 256) { float d = x[j] - mu; v += d * d; }
    red[t] = v; __syncthreads();
    for (int o = 128; o > 0; o >>= 1) { if (t < o) red[t] += red[t + o]; __syncthreads(); }
    float rstd = rsqrtf(red[0] * (1.0f / DMODEL) + 1e-5f);
    for (int j = t; j < DMODEL; j += 256)
        out[(size_t)m * DMODEL + j] = (x[j] - mu) * rstd * gamma[j] + beta[j];
}

// ---------------- host launcher --------------------------------------------
extern "C" void kernel_launch(void* const* d_in, const int* in_sizes, int n_in,
                              void* d_out, int out_size, void* d_ws, size_t ws_size,
                              hipStream_t stream)
{
    const float* h     = (const float*)d_in[0];
    const float* Wq    = (const float*)d_in[1];
    const float* Wkv   = (const float*)d_in[2];
    const float* Wo    = (const float*)d_in[3];
    const float* gamma = (const float*)d_in[4];
    const float* beta  = (const float*)d_in[5];
    float* out = (float*)d_out;
    (void)in_sizes; (void)n_in; (void)out_size; (void)ws_size;

    char* ws = (char*)d_ws;
    size_t off = 0;
    auto take = [&](size_t bytes) { void* p = ws + off; off += (bytes + 255) & ~(size_t)255; return p; };
    bf16*  hb   = (bf16*)take((size_t)MROWS * DMODEL * 2);        // 4 MB
    bf16*  Wqkv = (bf16*)take((size_t)3072 * DMODEL * 2);         // 6 MB
    bf16*  Wob  = (bf16*)take((size_t)DMODEL * DMODEL * 2);       // 2 MB
    bf16*  Q    = (bf16*)take((size_t)32 * SEQ * DHEAD * 2);      // 4 MB
    bf16*  Kc   = (bf16*)take((size_t)32 * SEQ * DHEAD * 2);      // 4 MB
    bf16*  Vt   = (bf16*)take((size_t)32 * SEQ * DHEAD * 2);      // 4 MB
    bf16*  P    = (bf16*)take((size_t)32 * SEQ * SEQ * 2);        // 64 MB
    bf16*  Pt   = (bf16*)take((size_t)32 * SEQ * SEQ * 2);        // 64 MB
    bf16*  Pn   = (bf16*)take((size_t)32 * SEQ * SEQ * 2);        // 64 MB
    bf16*  vec  = (bf16*)take((size_t)MROWS * DMODEL * 2);        // 4 MB
    float* xb   = (float*)take((size_t)MROWS * DMODEL * 4);       // 8 MB

    k_pack    <<<dim3(8192),       dim3(256), 0, stream>>>(h, Wq, Wkv, Wo, hb, Wqkv, Wob);
    k_gemm_qkv<<<dim3(24, 16),     dim3(256), 0, stream>>>(hb, Wqkv, Q, Kc, Vt);
    k_attn    <<<dim3(64, 32),     dim3(256), 16 * SEQ * 4 + 2048, stream>>>(Q, Kc, P, Pt);
    k_twice   <<<dim3(8, 8, 32),   dim3(256), 0, stream>>>(P, Pt, Pn);
    k_av      <<<dim3(1, 16, 32),  dim3(256), 0, stream>>>(Pn, Vt, vec);
    k_out     <<<dim3(8, 16),      dim3(256), 0, stream>>>(vec, Wob, h, xb);
    k_ln      <<<dim3(2048),       dim3(256), 0, stream>>>(xb, gamma, beta, out);
}
